// Qwen3MoeDecoderLayer_46909632807486
// MI455X (gfx1250) — compile-verified, matrix operand order checked by reading, not run on hardware
//
#include <hip/hip_runtime.h>
#include <hip/hip_bf16.h>
#include <math.h>

typedef __bf16 bf16;
typedef __attribute__((ext_vector_type(16))) __bf16 v16bf;
typedef __attribute__((ext_vector_type(8)))  __bf16 v8bf;
typedef __attribute__((ext_vector_type(4)))  __bf16 v4bf;
typedef __attribute__((ext_vector_type(8)))  float  v8f;
typedef __attribute__((ext_vector_type(4)))  float  v4f;

constexpr int T_   = 2048;
constexpr int HID_ = 2048;
constexpr int H_   = 32;
constexpr int HKV_ = 4;
constexpr int D_   = 128;
constexpr int E_   = 64;
constexpr int I_   = 768;
constexpr int TOPK_= 8;
constexpr int QKVN_= (H_ + 2*HKV_) * D_;   // 5120
constexpr float EPS_ = 1e-6f;

// ---------------- WMMA helpers (wave32, 16x16x32 bf16 -> f32) ----------------

__device__ inline v8f wmma_bf16(v16bf a, v16bf b, v8f c) {
  return __builtin_amdgcn_wmma_f32_16x16x32_bf16(false, a, false, b, (short)0, c, false, false);
}

// A-fragment: lane holds row (lane&15); element e<8 -> K = kh*8+e, e>=8 -> K = 16+kh*8+(e-8)
// rowp must already point at  Arow + kh*8  (16B-aligned); works for global or LDS pointers.
__device__ inline v16bf load_frag_contig(const bf16* rowp) {
  v8bf lo = *reinterpret_cast<const v8bf*>(rowp);
  v8bf hi = *reinterpret_cast<const v8bf*>(rowp + 16);
  v16bf a;
#pragma unroll
  for (int i = 0; i < 8; ++i) { a[i] = lo[i]; a[i + 8] = hi[i]; }
  return a;
}

// B-fragment from an LDS-staged 32xN bf16 k-slice (row = k, padded row length 136)
__device__ inline v16bf bfrag_from_lds(const bf16 (*sB)[136], int kh, int n) {
  v16bf b;
#pragma unroll
  for (int i = 0; i < 8; ++i) {
    b[i]     = sB[kh * 8 + i][n];
    b[i + 8] = sB[16 + kh * 8 + i][n];
  }
  return b;
}

// stage one 32x128 fp32 slice of B (row-major, ld = ldn) into LDS bf16 [32][136]
// 256 threads, each thread: 4x float4 load + packed cvt + b64 LDS store
__device__ inline void stage_b_slice(bf16 (*sB)[136], const float* Bsrc, int k0, int n0,
                                     int ldn, int tid) {
#pragma unroll
  for (int j = 0; j < 4; ++j) {
    int f  = tid + j * 256;        // 0..1023
    int kk = f >> 5;               // 0..31
    int c4 = f & 31;               // float4 index within 128 cols
    v4f val = *reinterpret_cast<const v4f*>(Bsrc + (size_t)(k0 + kk) * ldn + n0 + c4 * 4);
    v4bf q;
#pragma unroll
    for (int c = 0; c < 4; ++c) q[c] = (bf16)val[c];
    *reinterpret_cast<v4bf*>(&sB[kk][c4 * 4]) = q;
  }
}

// ---------------- elementwise / norm kernels ----------------

__global__ __launch_bounds__(256) void k_addrms(const float* hs, const float* resid,
                                                const float* w, float* res, bf16* xb) {
  int t = blockIdx.x, tid = threadIdx.x;
  size_t base = (size_t)t * HID_;
  float v[8], ss = 0.f;
#pragma unroll
  for (int i = 0; i < 8; ++i) {
    int c = tid + i * 256;
    float x = hs[base + c] + resid[base + c];
    v[i] = x; res[base + c] = x; ss += x * x;
  }
  __shared__ float sred[256];
  sred[tid] = ss; __syncthreads();
  for (int s = 128; s > 0; s >>= 1) { if (tid < s) sred[tid] += sred[tid + s]; __syncthreads(); }
  float rn = rsqrtf(sred[0] / (float)HID_ + EPS_);
#pragma unroll
  for (int i = 0; i < 8; ++i) {
    int c = tid + i * 256;
    xb[base + c] = (bf16)(v[i] * rn * w[c]);
  }
}

__global__ __launch_bounds__(256) void k_rms(const float* in, const float* w, bf16* xb) {
  int t = blockIdx.x, tid = threadIdx.x;
  size_t base = (size_t)t * HID_;
  float v[8], ss = 0.f;
#pragma unroll
  for (int i = 0; i < 8; ++i) { float x = in[base + tid + i * 256]; v[i] = x; ss += x * x; }
  __shared__ float sred[256];
  sred[tid] = ss; __syncthreads();
  for (int s = 128; s > 0; s >>= 1) { if (tid < s) sred[tid] += sred[tid + s]; __syncthreads(); }
  float rn = rsqrtf(sred[0] / (float)HID_ + EPS_);
#pragma unroll
  for (int i = 0; i < 8; ++i)
    xb[base + tid + i * 256] = (bf16)(v[i] * rn * w[tid + i * 256]);
}

// per-head q/k RMSNorm + RoPE, v pass-through; grid (T, H+2*HKV), 64 threads
__global__ __launch_bounds__(64) void k_qkvpost(const float* qkv, const float* qw, const float* kw,
                                                const int* pos, bf16* qb, bf16* kb, bf16* vb) {
  int t = blockIdx.x, slot = blockIdx.y, i = threadIdx.x;
  const float* src = qkv + (size_t)t * QKVN_ + (size_t)slot * D_;
  float x1 = src[i], x2 = src[i + 64];
  if (slot < H_ + HKV_) { // q or k: rms + rope
    __shared__ float sr[64];
    sr[i] = x1 * x1 + x2 * x2; __syncthreads();
    for (int s = 32; s > 0; s >>= 1) { if (i < s) sr[i] += sr[i + s]; __syncthreads(); }
    float rn = rsqrtf(sr[0] / (float)D_ + EPS_);
    const float* wv = (slot < H_) ? qw : kw;
    float y1 = x1 * rn * wv[i];
    float y2 = x2 * rn * wv[i + 64];
    float f = (float)pos[t] * __expf(-(float)i * 0.21586735246819354f); // ln(1e6)/64
    float c = __cosf(f), s = __sinf(f);
    float o1 = y1 * c - y2 * s;
    float o2 = y2 * c + y1 * s;
    if (slot < H_) {
      bf16* d = qb + ((size_t)t * H_ + slot) * D_;
      d[i] = (bf16)o1; d[i + 64] = (bf16)o2;
    } else {
      bf16* d = kb + ((size_t)t * HKV_ + (slot - H_)) * D_;
      d[i] = (bf16)o1; d[i + 64] = (bf16)o2;
    }
  } else {
    bf16* d = vb + ((size_t)t * HKV_ + (slot - H_ - HKV_)) * D_;
    d[i] = (bf16)x1; d[i + 64] = (bf16)x2;
  }
}

// ---------------- dense GEMM: C = A(bf16,MxK) @ B(f32->bf16,KxN) [+ add] ----------------
// block = 256 (8 waves); block tile M=64, N=128; LDS-staged A & B; grid (N/128, M/64)
__global__ __launch_bounds__(256) void k_gemm(const bf16* __restrict__ A,
                                              const float* __restrict__ B,
                                              float* __restrict__ C,
                                              const float* __restrict__ add,
                                              int N, int Kd) {
  __shared__ bf16 sA[64][40];    // padded: row stride 80B, distinct banks for b128 reads
  __shared__ bf16 sB[32][136];   // padded k-slice
  int tid = threadIdx.x;
  int lane = tid & 31, w = tid >> 5;
  int kh = lane >> 4, mr = lane & 15;
  int n0g = blockIdx.x * 128;
  int m0  = blockIdx.y * 64;
  int nloc = w * 16 + mr;

  v8f acc[4];
#pragma unroll
  for (int s = 0; s < 4; ++s) acc[s] = (v8f){0,0,0,0,0,0,0,0};

  int arow = tid >> 2, aseg = tid & 3;
  const bf16* aptr = A + (size_t)(m0 + arow) * Kd + aseg * 8;

  for (int k0 = 0; k0 < Kd; k0 += 32) {
    *reinterpret_cast<v8bf*>(&sA[arow][aseg * 8]) = *reinterpret_cast<const v8bf*>(aptr + k0);
    stage_b_slice(sB, B, k0, n0g, N, tid);
    if (k0 + 32 < Kd)
      __builtin_prefetch(B + (size_t)(k0 + 32 + (tid >> 3)) * N + n0g + (tid & 7) * 16, 0, 1);
    __syncthreads();
    v16bf bfrag = bfrag_from_lds(sB, kh, nloc);
#pragma unroll
    for (int sub = 0; sub < 4; ++sub) {
      v16bf afrag = load_frag_contig(&sA[sub * 16 + mr][0] + kh * 8);
      acc[sub] = wmma_bf16(afrag, bfrag, acc[sub]);
    }
    __syncthreads();
  }

  int col = n0g + nloc;
  if (add) {
#pragma unroll
    for (int sub = 0; sub < 4; ++sub)
#pragma unroll
      for (int r = 0; r < 8; ++r) {
        size_t idx = (size_t)(m0 + sub * 16 + r + 8 * kh) * N + col;
        C[idx] = acc[sub][r] + add[idx];
      }
  } else {
#pragma unroll
    for (int sub = 0; sub < 4; ++sub)
#pragma unroll
      for (int r = 0; r < 8; ++r) {
        size_t idx = (size_t)(m0 + sub * 16 + r + 8 * kh) * N + col;
        C[idx] = acc[sub][r];
      }
  }
}

// ---------------- flash attention: 1 wave per (head, 16-row q tile) ----------------
__global__ __launch_bounds__(32) void k_attn(const bf16* Q, const bf16* Kmat, const bf16* Vmat,
                                             bf16* O) {
  int lane = threadIdx.x & 31;
  int kh = lane >> 4, mr = lane & 15;
  int q0 = blockIdx.x * 16;
  int h  = blockIdx.y;
  int hkv = h / (H_ / HKV_);
  __shared__ bf16 smP[16][32];

  v16bf qf[4];
  const bf16* qrow = Q + ((size_t)(q0 + mr) * H_ + h) * D_;
#pragma unroll
  for (int kk = 0; kk < 4; ++kk) qf[kk] = load_frag_contig(qrow + kk * 32 + kh * 8);

  v8f oacc[8];
#pragma unroll
  for (int dt = 0; dt < 8; ++dt) oacc[dt] = (v8f){0,0,0,0,0,0,0,0};
  float rmax[8], rsum[8];
#pragma unroll
  for (int r = 0; r < 8; ++r) { rmax[r] = -1e30f; rsum[r] = 0.f; }
  const float sc = 0.08838834764831845f; // 128^-0.5

  for (int kb = 0; kb < q0 + 16; kb += 32) {
    v8f s0 = {0,0,0,0,0,0,0,0}, s1 = {0,0,0,0,0,0,0,0};
    int key0 = kb + mr, key1 = kb + 16 + mr;
    const bf16* kr0 = Kmat + ((size_t)key0 * HKV_ + hkv) * D_;
    const bf16* kr1 = Kmat + ((size_t)key1 * HKV_ + hkv) * D_;
#pragma unroll
    for (int kk = 0; kk < 4; ++kk) {
      v16bf b0 = load_frag_contig(kr0 + kk * 32 + kh * 8);
      v16bf b1 = load_frag_contig(kr1 + kk * 32 + kh * 8);
      s0 = wmma_bf16(qf[kk], b0, s0);
      s1 = wmma_bf16(qf[kk], b1, s1);
    }
    float p0[8], p1[8], mloc[8];
#pragma unroll
    for (int r = 0; r < 8; ++r) {
      int qrowi = q0 + r + 8 * kh;
      p0[r] = (key0 <= qrowi) ? s0[r] * sc : -1e30f;
      p1[r] = (key1 <= qrowi) ? s1[r] * sc : -1e30f;
      mloc[r] = fmaxf(p0[r], p1[r]);
    }
#pragma unroll
    for (int off = 1; off < 16; off <<= 1)
#pragma unroll
      for (int r = 0; r < 8; ++r) mloc[r] = fmaxf(mloc[r], __shfl_xor(mloc[r], off, 32));
    float corr[8];
#pragma unroll
    for (int r = 0; r < 8; ++r) {
      float nm = fmaxf(rmax[r], mloc[r]);
      corr[r] = __expf(rmax[r] - nm);
      rmax[r] = nm;
      p0[r] = __expf(p0[r] - nm);
      p1[r] = __expf(p1[r] - nm);
    }
    float ls[8];
#pragma unroll
    for (int r = 0; r < 8; ++r) ls[r] = p0[r] + p1[r];
#pragma unroll
    for (int off = 1; off < 16; off <<= 1)
#pragma unroll
      for (int r = 0; r < 8; ++r) ls[r] += __shfl_xor(ls[r], off, 32);
#pragma unroll
    for (int r = 0; r < 8; ++r) rsum[r] = rsum[r] * corr[r] + ls[r];
#pragma unroll
    for (int dt = 0; dt < 8; ++dt)
#pragma unroll
      for (int r = 0; r < 8; ++r) oacc[dt][r] *= corr[r];

#pragma unroll
    for (int r = 0; r < 8; ++r) {
      smP[r + 8 * kh][mr]      = (bf16)p0[r];
      smP[r + 8 * kh][16 + mr] = (bf16)p1[r];
    }
    __syncthreads();
    v16bf pf;
#pragma unroll
    for (int i = 0; i < 8; ++i) {
      pf[i]     = smP[mr][kh * 8 + i];
      pf[i + 8] = smP[mr][16 + kh * 8 + i];
    }
    __syncthreads();

#pragma unroll
    for (int dt = 0; dt < 8; ++dt) {
      v16bf vf;
#pragma unroll
      for (int i = 0; i < 8; ++i) {
        int kl0 = kh * 8 + i, kl1 = 16 + kh * 8 + i;
        vf[i]     = Vmat[((size_t)(kb + kl0) * HKV_ + hkv) * D_ + dt * 16 + mr];
        vf[i + 8] = Vmat[((size_t)(kb + kl1) * HKV_ + hkv) * D_ + dt * 16 + mr];
      }
      oacc[dt] = wmma_bf16(pf, vf, oacc[dt]);
    }
  }
  float inv[8];
#pragma unroll
  for (int r = 0; r < 8; ++r) inv[r] = 1.f / rsum[r];
#pragma unroll
  for (int dt = 0; dt < 8; ++dt)
#pragma unroll
    for (int r = 0; r < 8; ++r)
      O[(size_t)(q0 + r + 8 * kh) * (H_ * D_) + h * D_ + dt * 16 + mr] =
          (bf16)(oacc[dt][r] * inv[r]);
}

// ---------------- router: logits, softmax top-8, renormalize ----------------
__global__ __launch_bounds__(64) void k_router(const bf16* X, const float* Wg,
                                               int* ids, float* wts) {
  int t = blockIdx.x, e = threadIdx.x;
  const bf16* xr = X + (size_t)t * HID_;
  float acc = 0.f;
  for (int k = 0; k < HID_; ++k) acc += (float)xr[k] * Wg[(size_t)k * E_ + e];
  __shared__ float sl[64];
  sl[e] = acc; __syncthreads();
  if (e == 0) {
    bool taken[E_];
    for (int j = 0; j < E_; ++j) taken[j] = false;
    float bm[TOPK_]; int bi[TOPK_];
    for (int j = 0; j < TOPK_; ++j) {
      float best = -1e30f; int bidx = 0;
      for (int q = 0; q < E_; ++q)
        if (!taken[q] && sl[q] > best) { best = sl[q]; bidx = q; }
      taken[bidx] = true; bm[j] = best; bi[j] = bidx;
    }
    float mx = bm[0], s = 0.f, w[TOPK_];
    for (int j = 0; j < TOPK_; ++j) { w[j] = __expf(bm[j] - mx); s += w[j]; }
    float invs = 1.f / s;
    for (int j = 0; j < TOPK_; ++j) {
      ids[t * TOPK_ + j] = bi[j];
      wts[t * TOPK_ + j] = w[j] * invs;
    }
  }
}

__global__ void k_count(const int* ids, int* counts) {
  int i = blockIdx.x * blockDim.x + threadIdx.x;
  if (i < T_ * TOPK_) atomicAdd(&counts[ids[i]], 1);
}
__global__ void k_scan(const int* counts, int* offs, int* cursor) {
  if (blockIdx.x == 0 && threadIdx.x == 0) {
    int run = 0;
    for (int e = 0; e < E_; ++e) { offs[e] = run; cursor[e] = run; run += counts[e]; }
  }
}
__global__ void k_scatter(const int* ids, const float* wts, int* cursor,
                          int* btok, float* bw) {
  int i = blockIdx.x * blockDim.x + threadIdx.x;
  if (i < T_ * TOPK_) {
    int e = ids[i];
    int p = atomicAdd(&cursor[e], 1);
    btok[p] = i / TOPK_;
    bw[p] = wts[i];
  }
}

// ---------------- MoE gathered GEMMs (LDS-staged, M tile = 64) ----------------
// act = silu(x@Wg) * (x@Wu); grid (I/128, T/64, E)
__global__ __launch_bounds__(256) void k_moe_gu(const bf16* __restrict__ X,
                                                const float* __restrict__ Wgu,
                                                const int* btok, const int* offs,
                                                const int* cnts, bf16* act) {
  int e = blockIdx.z;
  int cnt = cnts[e];
  int m0 = blockIdx.y * 64;
  if (m0 >= cnt) return;
  __shared__ bf16 sA[64][40];
  __shared__ bf16 sG[32][136];
  __shared__ bf16 sU[32][136];
  int tid = threadIdx.x;
  int lane = tid & 31, w = tid >> 5;
  int kh = lane >> 4, mr = lane & 15;
  int j0g = blockIdx.x * 128;
  int nloc = w * 16 + mr;
  int off = offs[e];

  int arow = tid >> 2, aseg = tid & 3;
  int rl = m0 + arow; if (rl > cnt - 1) rl = cnt - 1;
  const bf16* aptr = X + (size_t)btok[off + rl] * HID_ + aseg * 8;
  const float* Bw = Wgu + (size_t)e * HID_ * (2 * I_);

  v8f cg[4], cu[4];
#pragma unroll
  for (int s = 0; s < 4; ++s) { cg[s] = (v8f){0,0,0,0,0,0,0,0}; cu[s] = (v8f){0,0,0,0,0,0,0,0}; }

  for (int k0 = 0; k0 < HID_; k0 += 32) {
    *reinterpret_cast<v8bf*>(&sA[arow][aseg * 8]) = *reinterpret_cast<const v8bf*>(aptr + k0);
    stage_b_slice(sG, Bw, k0, j0g,      2 * I_, tid);
    stage_b_slice(sU, Bw, k0, I_ + j0g, 2 * I_, tid);
    __syncthreads();
    v16bf bg = bfrag_from_lds(sG, kh, nloc);
    v16bf bu = bfrag_from_lds(sU, kh, nloc);
#pragma unroll
    for (int sub = 0; sub < 4; ++sub) {
      v16bf afrag = load_frag_contig(&sA[sub * 16 + mr][0] + kh * 8);
      cg[sub] = wmma_bf16(afrag, bg, cg[sub]);
      cu[sub] = wmma_bf16(afrag, bu, cu[sub]);
    }
    __syncthreads();
  }
#pragma unroll
  for (int sub = 0; sub < 4; ++sub)
#pragma unroll
    for (int r = 0; r < 8; ++r) {
      int rr = m0 + sub * 16 + r + 8 * kh;
      if (rr < cnt) {
        float g = cg[sub][r], u = cu[sub][r];
        float s = g / (1.f + __expf(-g)) * u;
        act[(size_t)(off + rr) * I_ + j0g + nloc] = (bf16)s;
      }
    }
}

// out[t] += weight * (act @ W_down[e]); grid (HID/128, T/64, E)
__global__ __launch_bounds__(256) void k_moe_down(const bf16* __restrict__ act,
                                                  const float* __restrict__ Wd,
                                                  const int* btok, const float* bw,
                                                  const int* offs, const int* cnts,
                                                  float* out) {
  int e = blockIdx.z;
  int cnt = cnts[e];
  int m0 = blockIdx.y * 64;
  if (m0 >= cnt) return;
  __shared__ bf16 sA[64][40];
  __shared__ bf16 sB[32][136];
  int tid = threadIdx.x;
  int lane = tid & 31, w = tid >> 5;
  int kh = lane >> 4, mr = lane & 15;
  int n0g = blockIdx.x * 128;
  int nloc = w * 16 + mr;
  int off = offs[e];

  int arow = tid >> 2, aseg = tid & 3;
  int rl = m0 + arow; if (rl > cnt - 1) rl = cnt - 1;
  const bf16* aptr = act + (size_t)(off + rl) * I_ + aseg * 8;
  const float* Bw = Wd + (size_t)e * I_ * HID_;

  v8f acc[4];
#pragma unroll
  for (int s = 0; s < 4; ++s) acc[s] = (v8f){0,0,0,0,0,0,0,0};

  for (int k0 = 0; k0 < I_; k0 += 32) {
    *reinterpret_cast<v8bf*>(&sA[arow][aseg * 8]) = *reinterpret_cast<const v8bf*>(aptr + k0);
    stage_b_slice(sB, Bw, k0, n0g, HID_, tid);
    __syncthreads();
    v16bf bfrag = bfrag_from_lds(sB, kh, nloc);
#pragma unroll
    for (int sub = 0; sub < 4; ++sub) {
      v16bf afrag = load_frag_contig(&sA[sub * 16 + mr][0] + kh * 8);
      acc[sub] = wmma_bf16(afrag, bfrag, acc[sub]);
    }
    __syncthreads();
  }
#pragma unroll
  for (int sub = 0; sub < 4; ++sub)
#pragma unroll
    for (int r = 0; r < 8; ++r) {
      int rr = m0 + sub * 16 + r + 8 * kh;
      if (rr < cnt) {
        int p = off + rr;
        atomicAdd(&out[(size_t)btok[p] * HID_ + n0g + nloc], bw[p] * acc[sub][r]);
      }
    }
}

// ---------------- host launcher ----------------
extern "C" void kernel_launch(void* const* d_in, const int* in_sizes, int n_in,
                              void* d_out, int out_size, void* d_ws, size_t ws_size,
                              hipStream_t stream) {
  (void)in_sizes; (void)n_in; (void)out_size; (void)ws_size;
  const int*   positions = (const int*)  d_in[0];
  const float* hidden    = (const float*)d_in[1];
  const float* residual  = (const float*)d_in[2];
  const float* in_ln_w   = (const float*)d_in[3];
  const float* W_qkv     = (const float*)d_in[4];
  const float* q_norm_w  = (const float*)d_in[5];
  const float* k_norm_w  = (const float*)d_in[6];
  const float* W_o       = (const float*)d_in[7];
  const float* post_ln_w = (const float*)d_in[8];
  const float* W_gate    = (const float*)d_in[9];
  const float* W_gu      = (const float*)d_in[10];
  const float* W_down    = (const float*)d_in[11];

  float* out  = (float*)d_out;                     // [T, HID] MoE output
  float* res2 = (float*)d_out + (size_t)T_ * HID_; // [T, HID] second output

  char* ws = (char*)d_ws;
  size_t o = 0;
  auto carve = [&](size_t bytes) { char* p = ws + o; o = (o + bytes + 255) & ~(size_t)255; return p; };
  float* res    = (float*)carve((size_t)T_ * HID_ * 4);
  bf16*  xb     = (bf16*) carve((size_t)T_ * HID_ * 2);
  float* qkv    = (float*)carve((size_t)T_ * QKVN_ * 4);
  bf16*  qb     = (bf16*) carve((size_t)T_ * H_ * D_ * 2);
  bf16*  kb     = (bf16*) carve((size_t)T_ * HKV_ * D_ * 2);
  bf16*  vb     = (bf16*) carve((size_t)T_ * HKV_ * D_ * 2);
  bf16*  ob     = (bf16*) carve((size_t)T_ * H_ * D_ * 2);
  bf16*  x2b    = (bf16*) carve((size_t)T_ * HID_ * 2);
  int*   ids    = (int*)  carve((size_t)T_ * TOPK_ * 4);
  float* wts    = (float*)carve((size_t)T_ * TOPK_ * 4);
  int*   counts = (int*)  carve(E_ * 4);
  int*   offs   = (int*)  carve(E_ * 4);
  int*   cursor = (int*)  carve(E_ * 4);
  int*   btok   = (int*)  carve((size_t)T_ * TOPK_ * 4);
  float* bw     = (float*)carve((size_t)T_ * TOPK_ * 4);
  bf16*  act    = (bf16*) carve((size_t)T_ * TOPK_ * I_ * 2);

  hipMemsetAsync(out, 0, (size_t)T_ * HID_ * sizeof(float), stream);
  hipMemsetAsync(counts, 0, E_ * sizeof(int), stream);

  k_addrms<<<T_, 256, 0, stream>>>(hidden, residual, in_ln_w, res, xb);
  k_gemm<<<dim3(QKVN_ / 128, T_ / 64), 256, 0, stream>>>(xb, W_qkv, qkv, nullptr, QKVN_, HID_);
  k_qkvpost<<<dim3(T_, H_ + 2 * HKV_), 64, 0, stream>>>(qkv, q_norm_w, k_norm_w, positions, qb, kb, vb);
  k_attn<<<dim3(T_ / 16, H_), 32, 0, stream>>>(qb, kb, vb, ob);
  k_gemm<<<dim3(HID_ / 128, T_ / 64), 256, 0, stream>>>(ob, W_o, res2, res, HID_, H_ * D_);
  k_rms<<<T_, 256, 0, stream>>>(res2, post_ln_w, x2b);
  k_router<<<T_, 64, 0, stream>>>(x2b, W_gate, ids, wts);
  k_count<<<(T_ * TOPK_ + 255) / 256, 256, 0, stream>>>(ids, counts);
  k_scan<<<1, 32, 0, stream>>>(counts, offs, cursor);
  k_scatter<<<(T_ * TOPK_ + 255) / 256, 256, 0, stream>>>(ids, wts, cursor, btok, bw);
  k_moe_gu<<<dim3(I_ / 128, T_ / 64, E_), 256, 0, stream>>>(x2b, W_gu, btok, offs, counts, act);
  k_moe_down<<<dim3(HID_ / 128, T_ / 64, E_), 256, 0, stream>>>(act, W_down, btok, bw, offs, counts, out);
}
